// LinearAttention_MLP_35304631174207
// MI455X (gfx1250) — matrix operand order repaired and checked by reading, not compiled, not run on hardware
//
#include <hip/hip_runtime.h>

typedef __attribute__((ext_vector_type(16))) _Float16 v16h;
typedef __attribute__((ext_vector_type(8)))  _Float16 v8h;
typedef __attribute__((ext_vector_type(4)))  _Float16 v4h;
typedef __attribute__((ext_vector_type(8)))  float    v8f;
typedef __attribute__((ext_vector_type(4)))  int      i4;

union Frag { v16h v; v8h h[2]; };

#define DIM      1024
#define HIDDEN   512
#define HEADS    8
#define BATCH    16384
#define SCALE    0.125f
#define SQRTD    32.0f

typedef __attribute__((address_space(1))) i4 gi4;
typedef __attribute__((address_space(3))) i4 li4;

// 16B global -> LDS async copy (gfx1250 GLOBAL_LOAD_ASYNC_TO_LDS_B128)
__device__ __forceinline__ void async_cp16(const _Float16* g, _Float16* l) {
  i4* gg = (i4*)g;
  i4* ll = (i4*)l;
  __builtin_amdgcn_global_load_async_to_lds_b128((gi4*)gg, (li4*)ll, 0, 0);
}

// ---------------------------------------------------------------------------
// Prepass: x f32 -> f16, and weights -> f16 (wqv packed head-paired).
// ---------------------------------------------------------------------------
__global__ void k_pack_x(const float* __restrict__ x, _Float16* __restrict__ o) {
  size_t idx = (size_t)blockIdx.x * 256 + threadIdx.x;   // one float4 each
  const float4 v = *(const float4*)(x + idx * 4);
  v4h h = {(_Float16)v.x, (_Float16)v.y, (_Float16)v.z, (_Float16)v.w};
  *(v4h*)(o + idx * 4) = h;
}

__global__ void k_pack_wqv(const float* __restrict__ w, _Float16* __restrict__ o) {
  int idx = blockIdx.x * 256 + threadIdx.x;       // 0 .. 1024*1024-1
  int i = idx >> 10, k = idx & 1023;
  int head = i >> 7, j = i & 127;
  int src = (j < 64) ? (head * 64 + j) : (2 * HIDDEN + head * 64 + (j - 64));
  o[idx] = (_Float16)w[(size_t)src * DIM + k];
}

__global__ void k_pack_wout(const float* __restrict__ w, _Float16* __restrict__ o) {
  int idx = blockIdx.x * 256 + threadIdx.x;       // 0 .. 1024*512-1
  o[idx] = (_Float16)w[idx];
}

// ---------------------------------------------------------------------------
// Kernel A: fused (x @ wqv^T) for one head's q+v columns + attention epilogue
// Tile: BM=128 rows x 128 cols (q 0..63 | v 64..127), K=1024, BK=64.
// Double-buffered LDS filled by async-to-LDS DMA; 8 waves.
// ---------------------------------------------------------------------------
#define BM    128
#define BK    64
#define LDA   72            // halves per LDS row: 144B = 9*16B
#define STAGE (2 * BM * LDA * 2)   // A+B per stage = 36864 B
#define LDQ   66            // floats per epilogue q row
#define LDV   66            // halves per epilogue v row

__global__ __launch_bounds__(256) void ka_qkv_attn(
    const _Float16* __restrict__ xh, const _Float16* __restrict__ wqv,
    _Float16* __restrict__ ov)
{
  // K-loop: two stages of (A 18432 + B 18432) = 73728 B
  // epilogue reuse: sEq 33792 + sEv 16896 = 50688 B (fits in same region)
  __shared__ __align__(16) unsigned char smem[2 * STAGE];
  float*    sEq = (float*)smem;
  _Float16* sEv = (_Float16*)(smem + BM * LDQ * 4);
  __shared__ float sS[BM];

  const int tid  = threadIdx.x;
  const int wave = tid >> 5;
  const int lane = tid & 31;
  const int rl   = lane & 15;
  const int sel  = lane >> 4;

  const _Float16* xb = xh  + (size_t)blockIdx.x * BM * DIM;
  const _Float16* wb = wqv + (size_t)blockIdx.y * 128 * DIM;

  v8f zero = {0.f, 0.f, 0.f, 0.f, 0.f, 0.f, 0.f, 0.f};
  v8f acc[8];
#pragma unroll
  for (int n = 0; n < 8; n++) acc[n] = zero;

  // issue one stage: 8 async b128 per wave (4 for A tile, 4 for B tile)
  auto issueStage = [&](int kt, int p) {
    const _Float16* xg = xb + kt * BK;
    const _Float16* wg = wb + kt * BK;
    _Float16* lA = (_Float16*)(smem + p * STAGE);
    _Float16* lB = lA + BM * LDA;
#pragma unroll
    for (int i = 0; i < 4; i++) {
      int idx = tid + i * 256;          // 0..1023 ; 8 chunks of 16B per row
      int r = idx >> 3, c = idx & 7;
      async_cp16(xg + (size_t)r * DIM + c * 8, lA + r * LDA + c * 8);
    }
#pragma unroll
    for (int i = 0; i < 4; i++) {
      int idx = tid + i * 256;
      int r = idx >> 3, c = idx & 7;
      async_cp16(wg + (size_t)r * DIM + c * 8, lB + r * LDA + c * 8);
    }
  };

  issueStage(0, 0);

  for (int kt = 0; kt < DIM / BK; kt++) {
    const int cur = kt & 1;
    __syncthreads();                    // buf(1-cur) free: all waves done kt-1
    if (kt + 1 < DIM / BK) {
      issueStage(kt + 1, 1 - cur);
      __builtin_amdgcn_s_wait_asynccnt(8);   // stage kt landed (in-order)
    } else {
      __builtin_amdgcn_s_wait_asynccnt(0);
    }
    __syncthreads();                    // all waves' stage-kt data visible

    const _Float16* cA = (const _Float16*)(smem + cur * STAGE);
    const _Float16* cB = cA + BM * LDA;

    Frag a0, a1;
    {
      const _Float16* pa = cA + (wave * 16 + rl) * LDA + sel * 8;
      a0.h[0] = *(const v8h*)pa;        // kh=0: elems = K sel*8.. , sel*8+16..
      a0.h[1] = *(const v8h*)(pa + 16);
      a1.h[0] = *(const v8h*)(pa + 32); // kh=1
      a1.h[1] = *(const v8h*)(pa + 48);
    }
#pragma unroll
    for (int n = 0; n < 8; n++) {
      const _Float16* pb = cB + (n * 16 + rl) * LDA + sel * 8;
      Frag b0, b1;
      b0.h[0] = *(const v8h*)pb;        // kh=0: elems = K sel*8.. , sel*8+16..
      b0.h[1] = *(const v8h*)(pb + 16);
      b1.h[0] = *(const v8h*)(pb + 32); // kh=1
      b1.h[1] = *(const v8h*)(pb + 48);
      acc[n] = __builtin_amdgcn_wmma_f32_16x16x32_f16(
          false, a0.v, false, b0.v, (short)0, acc[n], false, false);
      acc[n] = __builtin_amdgcn_wmma_f32_16x16x32_f16(
          false, a1.v, false, b1.v, (short)0, acc[n], false, false);
    }
  }

  // ---- epilogue: softmax-sum over q (64 cols) per row, then s * v ----
  __syncthreads();                      // done with staging buffers
  {
    int rbase = wave * 16 + sel * 8;
#pragma unroll
    for (int n = 0; n < 4; n++)
#pragma unroll
      for (int r = 0; r < 8; r++)
        sEq[(rbase + r) * LDQ + n * 16 + rl] = acc[n][r];
#pragma unroll
    for (int n = 4; n < 8; n++)
#pragma unroll
      for (int r = 0; r < 8; r++)
        sEv[(rbase + r) * LDV + (n - 4) * 16 + rl] = (_Float16)acc[n][r];
  }
  __syncthreads();

  if (tid < BM) {
    const float* q = sEq + tid * LDQ;
    float mx = q[0];
#pragma unroll
    for (int d = 1; d < 64; d++) mx = fmaxf(mx, q[d]);
    float Z = 0.f;
#pragma unroll
    for (int d = 0; d < 64; d++) Z += __expf(q[d] - mx);
    float s = 0.f;
#pragma unroll
    for (int d = 0; d < 64; d++) s += __expf(q[d] - mx) / Z;  // == Σ softmax
    sS[tid] = s * SCALE;
  }
  __syncthreads();

  _Float16* ovb = ov + (size_t)blockIdx.x * BM * HIDDEN + blockIdx.y * 64;
#pragma unroll
  for (int i = 0; i < 32; i++) {
    int idx = tid + i * 256;            // 0..8191
    int r = idx >> 6, c = idx & 63;
    float val = sS[r] * (float)sEv[r * LDV + c];
    ovb[(size_t)r * HIDDEN + c] = (_Float16)val;
  }
}

// ---------------------------------------------------------------------------
// Kernel B: out = ov @ w_out^T + b, fused row RMS-norm * g * 32.
// Tile: 32 rows x full 1024 cols, K=512. A panel async-DMA'd to LDS once;
// B fragments streamed from L2-resident w_out. Wave w: rows (w&1)*16,
// cols (w>>1)*256 (16 C tiles each).
// ---------------------------------------------------------------------------
#define BM2  32
#define LDA2 520    // halves: 1040B = 65*16B

__global__ __launch_bounds__(256) void kb_out_norm(
    const _Float16* __restrict__ ov, const _Float16* __restrict__ wo,
    const float* __restrict__ bias, const float* __restrict__ g,
    float* __restrict__ out)
{
  __shared__ __align__(16) _Float16 sA[BM2 * LDA2];
  __shared__ float sSS[BM2];

  const int tid  = threadIdx.x;
  const int wave = tid >> 5;
  const int lane = tid & 31;
  const int rl   = lane & 15;
  const int sel  = lane >> 4;

  const _Float16* ab = ov + (size_t)blockIdx.x * BM2 * HIDDEN;

#pragma unroll
  for (int i = 0; i < 8; i++) {
    int idx = tid + i * 256;            // 0..2047 chunks of 16B
    int r = idx >> 6, c8 = idx & 63;
    async_cp16(ab + (size_t)r * HIDDEN + c8 * 8, sA + r * LDA2 + c8 * 8);
  }
  if (tid < BM2) sSS[tid] = 0.f;
  __builtin_amdgcn_s_wait_asynccnt(0);
  __syncthreads();

  const int rslab = (wave & 1) * 16;
  const int cslab = (wave >> 1) * 256;

  v8f zero = {0.f, 0.f, 0.f, 0.f, 0.f, 0.f, 0.f, 0.f};
  v8f acc[16];
#pragma unroll
  for (int n = 0; n < 16; n++) acc[n] = zero;

  for (int kt = 0; kt < HIDDEN / 32; kt++) {
    Frag a;
    const _Float16* pa = sA + (rslab + rl) * LDA2 + kt * 32 + sel * 8;
    a.h[0] = *(const v8h*)pa;
    a.h[1] = *(const v8h*)(pa + 16);
#pragma unroll
    for (int n = 0; n < 16; n++) {
      Frag b;
      const _Float16* pb = wo + (size_t)(cslab + n * 16 + rl) * HIDDEN + kt * 32 + sel * 8;
      b.h[0] = *(const v8h*)pb;
      b.h[1] = *(const v8h*)(pb + 16);
      acc[n] = __builtin_amdgcn_wmma_f32_16x16x32_f16(
          false, a.v, false, b.v, (short)0, acc[n], false, false);
    }
  }

  // ---- bias + per-row sum of squares ----
  float ss[8];
#pragma unroll
  for (int r = 0; r < 8; r++) ss[r] = 0.f;
#pragma unroll
  for (int n = 0; n < 16; n++) {
    float bo = bias[cslab + n * 16 + rl];
#pragma unroll
    for (int r = 0; r < 8; r++) {
      float v = acc[n][r] + bo;
      acc[n][r] = v;
      ss[r] += v * v;
    }
  }
#pragma unroll
  for (int r = 0; r < 8; r++) {
    float v = ss[r];
    v += __shfl_xor(v, 1, 32);
    v += __shfl_xor(v, 2, 32);
    v += __shfl_xor(v, 4, 32);
    v += __shfl_xor(v, 8, 32);
    ss[r] = v;                          // half-wave holds its rows' sums
  }
  const int rbase = rslab + sel * 8;
  if (rl == 0) {
#pragma unroll
    for (int r = 0; r < 8; r++) atomicAdd(&sSS[rbase + r], ss[r]);
  }
  __syncthreads();

  float inv[8];
#pragma unroll
  for (int r = 0; r < 8; r++) {
    float nrm = sqrtf(sSS[rbase + r]);
    inv[r] = SQRTD / fmaxf(nrm, 1e-12f);
  }

  float* ob = out + (size_t)blockIdx.x * BM2 * DIM;
#pragma unroll
  for (int n = 0; n < 16; n++) {
    int col = cslab + n * 16 + rl;
    float gs = g[col];
#pragma unroll
    for (int r = 0; r < 8; r++)
      ob[(size_t)(rbase + r) * DIM + col] = acc[n][r] * gs * inv[r];
  }
}

// ---------------------------------------------------------------------------
extern "C" void kernel_launch(void* const* d_in, const int* in_sizes, int n_in,
                              void* d_out, int out_size, void* d_ws, size_t ws_size,
                              hipStream_t stream) {
  (void)in_sizes; (void)n_in; (void)out_size; (void)ws_size;
  const float* x     = (const float*)d_in[0];
  const float* w_qkv = (const float*)d_in[1];
  const float* w_out = (const float*)d_in[2];
  const float* b_out = (const float*)d_in[3];
  const float* g     = (const float*)d_in[4];
  float* out = (float*)d_out;

  char* ws = (char*)d_ws;
  _Float16* ov   = (_Float16*)ws;                          // 16 MB
  _Float16* wqvh = (_Float16*)(ws + 16u * 1024 * 1024);    //  2 MB
  _Float16* woh  = (_Float16*)(ws + 18u * 1024 * 1024);    //  1 MB
  _Float16* xh   = (_Float16*)(ws + 20u * 1024 * 1024);    // 32 MB

  k_pack_x<<<(BATCH * DIM / 4) / 256, 256, 0, stream>>>(x, xh);
  k_pack_wqv<<<(1024 * 1024) / 256, 256, 0, stream>>>(w_qkv, wqvh);
  k_pack_wout<<<(1024 * 512) / 256, 256, 0, stream>>>(w_out, woh);
  ka_qkv_attn<<<dim3(BATCH / BM, HEADS), 256, 0, stream>>>(xh, wqvh, ov);
  kb_out_norm<<<BATCH / BM2, 256, 0, stream>>>(ov, woh, b_out, g, out);
}
